// SSMBlock_2276332667655
// MI455X (gfx1250) — matrix operand order repaired and checked by reading, not compiled, not run on hardware
//
#include <hip/hip_runtime.h>
#include <hip/hip_bf16.h>
#include <math.h>

// ---------------------------------------------------------------------------
// Types matching gfx1250 WMMA builtin signatures (probe-confirmed):
//   __builtin_amdgcn_wmma_f32_16x16x32_bf16(bool, v16bf, bool, v16bf, short, v8f, bool, bool)
// ---------------------------------------------------------------------------
typedef __attribute__((ext_vector_type(16))) __bf16 bf16x16;
typedef __attribute__((ext_vector_type(8)))  __bf16 bf16x8;
typedef __attribute__((ext_vector_type(8)))  float  f32x8;

#define BTOT 2
#define LTOT 1024
#define FTOT 1024
#define MROWS 2048          // B*L
#define NSTATE 16
#define NDH 64
#define EPS 1e-5f

// ---------------------------------------------------------------------------
// WMMA fragment loaders.
// 16-bit A matrix 16x32 (ISA 7.12.2): lane = half*16 + m; elements e<8 hold
// K = half*8+e, e>=8 hold K = 16+half*8+(e-8)  -> two contiguous 16B loads.
// ---------------------------------------------------------------------------
__device__ __forceinline__ bf16x16 load_a_frag(const __bf16* A, int lda,
                                               int row, int k0, int half) {
  const __bf16* p = A + (size_t)row * lda + k0 + half * 8;
  bf16x8 lo = *(const bf16x8*)(p);
  bf16x8 hi = *(const bf16x8*)(p + 16);
  bf16x16 r;
#pragma unroll
  for (int i = 0; i < 8; ++i) { r[i] = lo[i]; r[8 + i] = hi[i]; }
  return r;
}

// 16-bit B matrix 32x16 (K x N), element(K,n) taken from W[n][K] (i.e. W^T):
// lane = half*16 + n; elements e<8 hold K = half*8+e, e>=8 hold K = 16+half*8+(e-8).
__device__ __forceinline__ bf16x16 load_b_fragT(const __bf16* W, int ldw,
                                                int col, int k0, int half) {
  const __bf16* p = W + (size_t)col * ldw + k0 + half * 8;
  bf16x8 lo = *(const bf16x8*)(p);
  bf16x8 hi = *(const bf16x8*)(p + 16);
  bf16x16 r;
#pragma unroll
  for (int i = 0; i < 8; ++i) { r[i] = lo[i]; r[8 + i] = hi[i]; }
  return r;
}

// ---------------------------------------------------------------------------
// Small utility kernels
// ---------------------------------------------------------------------------
__global__ void k_zero(float* p, int n) {
  int i = blockIdx.x * 256 + threadIdx.x;
  if (i < n) p[i] = 0.f;
}

__global__ void k_f2b(const float* s, __bf16* d, int n) {
  int i = blockIdx.x * 256 + threadIdx.x;
  if (i < n) d[i] = (__bf16)s[i];
}

// conv_w (FI, FI, 3) -> three FIxFI bf16 tap matrices
__global__ void k_convw(const float* s, __bf16* d0, __bf16* d1, __bf16* d2) {
  int i = blockIdx.x * 256 + threadIdx.x;   // 1M
  d0[i] = (__bf16)s[3 * i + 0];
  d1[i] = (__bf16)s[3 * i + 1];
  d2[i] = (__bf16)s[3 * i + 2];
}

// ---------------------------------------------------------------------------
// GroupNorm(1) over (L,F) per batch: sum/sumsq reduction
// ---------------------------------------------------------------------------
__global__ void k_batch_stats(const float* x, float* stats) {
  __shared__ float ss[256], sq[256];
  int b = blockIdx.x >> 6;          // 64 blocks per batch
  int blk = blockIdx.x & 63;
  size_t base = (size_t)b * (LTOT * FTOT) + (size_t)blk * 16384;
  float s = 0.f, q = 0.f;
  for (int i = threadIdx.x; i < 16384; i += 256) {
    float v = x[base + i];
    s += v; q += v * v;
  }
  ss[threadIdx.x] = s; sq[threadIdx.x] = q;
  __syncthreads();
  for (int o = 128; o > 0; o >>= 1) {
    if (threadIdx.x < o) { ss[threadIdx.x] += ss[threadIdx.x + o];
                           sq[threadIdx.x] += sq[threadIdx.x + o]; }
    __syncthreads();
  }
  if (threadIdx.x == 0) {
    atomicAdd(&stats[b],     ss[0]);
    atomicAdd(&stats[2 + b], sq[0]);
  }
}

__global__ void k_xnorm(const float* x, const float* stats, __bf16* xn) {
  size_t i = (size_t)blockIdx.x * 256 + threadIdx.x;   // 2M
  int b = (int)(i >> 20);
  const float inv = 1.f / (float)(LTOT * FTOT);
  float mean = stats[b] * inv;
  float var  = stats[2 + b] * inv - mean * mean;
  float rstd = rsqrtf(var + EPS);
  xn[i] = (__bf16)((x[i] - mean) * rstd);
}

// ---------------------------------------------------------------------------
// Register-blocked WMMA GEMM: OUT[M,N] = A[M,K] bf16 @ W[N,K]^T bf16 + bias,
// f32 accumulate. Each wave owns a 16 x (16*NT) strip: the A fragment is
// loaded once per k-step and reused against NT B fragments -> 2+2*NT b128
// loads per NT WMMAs (2.5 loads/WMMA at NT=4 vs 4 without blocking).
// mode 0: f32 out; 1: bf16 out; 2: f32 + softplus; 3: f32 + residual;
// mode 4: f32 out + bf16 copy.
// ---------------------------------------------------------------------------
template <int NT>
__global__ void k_wmma_gemm(const __bf16* __restrict__ A,
                            const __bf16* __restrict__ W,
                            const float* __restrict__ bias,
                            float* outF, __bf16* outB,
                            const float* __restrict__ resid,
                            int M, int N, int K, int mode) {
  int wave = threadIdx.x >> 5;
  int lane = threadIdx.x & 31;
  int stripsN = N / (16 * NT);
  int tilesM  = M >> 4;
  int strip = blockIdx.x * 8 + wave;
  if (strip >= tilesM * stripsN) return;     // uniform per wave: EXEC stays full
  int tm = strip / stripsN, ts = strip % stripsN;
  int half = lane >> 4;
  int mrow = tm * 16 + (lane & 15);
  int n0   = ts * 16 * NT + (lane & 15);
  f32x8 acc[NT];
#pragma unroll
  for (int j = 0; j < NT; ++j) acc[j] = (f32x8){};
  for (int k0 = 0; k0 < K; k0 += 32) {
    bf16x16 af = load_a_frag(A, K, mrow, k0, half);
#pragma unroll
    for (int j = 0; j < NT; ++j) {
      bf16x16 bf_ = load_b_fragT(W, K, n0 + j * 16, k0, half);
      acc[j] = __builtin_amdgcn_wmma_f32_16x16x32_bf16(false, af, false, bf_,
                                                       (short)0, acc[j], false, false);
    }
  }
#pragma unroll
  for (int j = 0; j < NT; ++j) {
    int ncol = n0 + j * 16;
    float bv = bias[ncol];
#pragma unroll
    for (int r = 0; r < 8; ++r) {
      int m = tm * 16 + half * 8 + r;        // D layout: lanes 0-15 M=r, 16-31 M=8+r
      size_t idx = (size_t)m * N + ncol;
      float v = acc[j][r] + bv;
      if (mode == 2) v = (v > 20.f) ? v : log1pf(__expf(v));   // softplus
      if (mode == 3) v += resid[idx];
      if (mode == 1) {
        outB[idx] = (__bf16)v;
      } else {
        outF[idx] = v;
        if (mode == 4) outB[idx] = (__bf16)v;
      }
    }
  }
}

// ---------------------------------------------------------------------------
// Conv1d(k=3, pad=1) over channels as three shifted WMMA GEMMs, register-
// blocked 16x64 per wave (per tap: A loaded once, reused for 4 N-tiles).
// Per-channel bias omitted: InstanceNorm over L cancels it exactly.
// ---------------------------------------------------------------------------
__global__ void k_conv_wmma(const __bf16* __restrict__ H,
                            const __bf16* __restrict__ W0,
                            const __bf16* __restrict__ W1,
                            const __bf16* __restrict__ W2,
                            float* out) {
  int wave = threadIdx.x >> 5;
  int lane = threadIdx.x & 31;
  const int stripsN = FTOT / 64;             // 16 strips of 4 tiles
  int strip = blockIdx.x * 8 + wave;         // 128 * 16 = 2048 strips
  int tm = strip / stripsN, ts = strip % stripsN;
  int half = lane >> 4;
  int g = tm * 16 + (lane & 15);             // flattened (b,l) row
  int l = g & (LTOT - 1);
  int n0 = ts * 64 + (lane & 15);
  const __bf16* Wt[3] = {W0, W1, W2};
  f32x8 acc[4];
#pragma unroll
  for (int j = 0; j < 4; ++j) acc[j] = (f32x8){};
  for (int k0 = 0; k0 < FTOT; k0 += 32) {
#pragma unroll
    for (int t = 0; t < 3; ++t) {
      bf16x16 af;
      int ls = l + t - 1;
      if (ls < 0 || ls >= LTOT) {
#pragma unroll
        for (int i = 0; i < 16; ++i) af[i] = (__bf16)0.f;   // zero-pad row
      } else {
        af = load_a_frag(H, FTOT, g + t - 1, k0, half);
      }
#pragma unroll
      for (int j = 0; j < 4; ++j) {
        bf16x16 bf_ = load_b_fragT(Wt[t], FTOT, n0 + j * 16, k0, half);
        acc[j] = __builtin_amdgcn_wmma_f32_16x16x32_bf16(false, af, false, bf_,
                                                         (short)0, acc[j], false, false);
      }
    }
  }
#pragma unroll
  for (int j = 0; j < 4; ++j) {
#pragma unroll
    for (int r = 0; r < 8; ++r) {
      int m = tm * 16 + half * 8 + r;
      out[(size_t)m * FTOT + n0 + j * 16] = acc[j][r];
    }
  }
}

// ---------------------------------------------------------------------------
// InstanceNorm over L per (b, channel): stats then normalize + SiLU
// ---------------------------------------------------------------------------
__global__ void k_chan_stats(const float* hc, float* csum, float* csq) {
  // grid = 2 (b) * 16 (l-chunks) * 4 (channel blocks) = 128
  int b = blockIdx.x >> 6;
  int lchunk = (blockIdx.x >> 2) & 15;
  int cblk = blockIdx.x & 3;
  int c = cblk * 256 + threadIdx.x;
  float s = 0.f, q = 0.f;
  for (int j = 0; j < 64; ++j) {
    int l = lchunk * 64 + j;
    float v = hc[((size_t)b * LTOT + l) * FTOT + c];
    s += v; q += v * v;
  }
  atomicAdd(&csum[b * FTOT + c], s);
  atomicAdd(&csq[b * FTOT + c], q);
}

__global__ void k_chan_norm_silu(const float* hc, const float* csum,
                                 const float* csq, float* hp, __bf16* hpb) {
  size_t i = (size_t)blockIdx.x * 256 + threadIdx.x;   // 2M
  int c = (int)(i & (FTOT - 1));
  int b = (int)(i >> 20);
  float mean = csum[b * FTOT + c] * (1.f / LTOT);
  float var  = csq[b * FTOT + c] * (1.f / LTOT) - mean * mean;
  float rstd = rsqrtf(var + EPS);
  float v = (hc[i] - mean) * rstd;
  v = v / (1.f + __expf(-v));                // SiLU
  hp[i] = v;
  hpb[i] = (__bf16)v;
}

// ---------------------------------------------------------------------------
// Selective scan: 32768 independent (b,f,n) recurrences, one per thread.
// s_n <- exp(-Delta*A[f,n]) * s_n + (Delta*h)*B_n ;  y = sum_n s_n*C_n + D*h
// y-reduction across the 16 n-lanes via shfl_xor.
// ---------------------------------------------------------------------------
__global__ void k_ssm_scan(const float* __restrict__ Delta,
                           const float* __restrict__ h,
                           const float* __restrict__ Bm,
                           const float* __restrict__ Cm,
                           const float* __restrict__ A_log,
                           const float* __restrict__ Dvec,
                           float* __restrict__ y) {
  int t = blockIdx.x * 256 + threadIdx.x;    // 32768 threads
  int n = t & 15;
  int f = (t >> 4) & (FTOT - 1);
  int b = t >> 14;
  float a = __expf(A_log[f * NSTATE + n]);
  float dco = Dvec[f];
  float s = 0.f;
  size_t base_lf = (size_t)b * LTOT * FTOT + f;
  size_t base_ln = (size_t)b * LTOT * NSTATE + n;
  for (int l = 0; l < LTOT; ++l) {
    float d  = Delta[base_lf + (size_t)l * FTOT];
    float hv = h[base_lf + (size_t)l * FTOT];
    float bn = Bm[base_ln + (size_t)l * NSTATE];
    float cn = Cm[base_ln + (size_t)l * NSTATE];
    s = __expf(-d * a) * s + d * hv * bn;
    float yp = s * cn;
    yp += __shfl_xor(yp, 1, 16);
    yp += __shfl_xor(yp, 2, 16);
    yp += __shfl_xor(yp, 4, 16);
    yp += __shfl_xor(yp, 8, 16);
    if (n == 0) y[base_lf + (size_t)l * FTOT] = yp + dco * hv;
  }
}

// ---------------------------------------------------------------------------
// g = y * silu(h0); row-norm over F per (b,l); emit bf16 for the out-proj GEMM
// ---------------------------------------------------------------------------
__global__ void k_gate_rownorm(const float* y, const float* h0, __bf16* gn) {
  __shared__ float ss[256], sq[256];
  __shared__ float smean, srstd;
  size_t base = (size_t)blockIdx.x * FTOT;
  float g[4];
  float s = 0.f, q = 0.f;
#pragma unroll
  for (int j = 0; j < 4; ++j) {
    size_t idx = base + threadIdx.x + j * 256;
    float hv = h0[idx];
    float gv = y[idx] * (hv / (1.f + __expf(-hv)));
    g[j] = gv; s += gv; q += gv * gv;
  }
  ss[threadIdx.x] = s; sq[threadIdx.x] = q;
  __syncthreads();
  for (int o = 128; o > 0; o >>= 1) {
    if (threadIdx.x < o) { ss[threadIdx.x] += ss[threadIdx.x + o];
                           sq[threadIdx.x] += sq[threadIdx.x + o]; }
    __syncthreads();
  }
  if (threadIdx.x == 0) {
    float mean = ss[0] * (1.f / FTOT);
    float var  = sq[0] * (1.f / FTOT) - mean * mean;
    smean = mean; srstd = rsqrtf(var + EPS);
  }
  __syncthreads();
#pragma unroll
  for (int j = 0; j < 4; ++j) {
    size_t idx = base + threadIdx.x + j * 256;
    gn[idx] = (__bf16)((g[j] - smean) * srstd);
  }
}

// ---------------------------------------------------------------------------
// Orchestration
// ---------------------------------------------------------------------------
extern "C" void kernel_launch(void* const* d_in, const int* in_sizes, int n_in,
                              void* d_out, int out_size, void* d_ws, size_t ws_size,
                              hipStream_t stream) {
  (void)in_sizes; (void)n_in; (void)out_size; (void)ws_size;
  const float* x      = (const float*)d_in[0];
  const float* w_in   = (const float*)d_in[1];
  const float* b_in   = (const float*)d_in[2];
  const float* conv_w = (const float*)d_in[3];
  /* d_in[4] conv_b unused: per-channel bias cancels exactly in InstanceNorm */
  const float* A_log  = (const float*)d_in[5];
  const float* Dvec   = (const float*)d_in[6];
  const float* w_B    = (const float*)d_in[7];
  const float* b_B    = (const float*)d_in[8];
  const float* w_C    = (const float*)d_in[9];
  const float* b_C    = (const float*)d_in[10];
  const float* w_d1   = (const float*)d_in[11];
  const float* b_d1   = (const float*)d_in[12];
  const float* w_d2   = (const float*)d_in[13];
  const float* b_d2   = (const float*)d_in[14];
  const float* w_out  = (const float*)d_in[15];
  const float* b_out  = (const float*)d_in[16];
  float* out = (float*)d_out;

  char* ws = (char*)d_ws;
  size_t off = 0;
  auto alloc = [&](size_t bytes) {
    size_t o = off; off += (bytes + 255) & ~(size_t)255; return o;
  };
  float*  stats  = (float*)(ws + alloc(16 * 4));
  float*  csum   = (float*)(ws + alloc(MROWS * 4));
  float*  csq    = (float*)(ws + alloc(MROWS * 4));
  __bf16* xn_bf  = (__bf16*)(ws + alloc((size_t)MROWS * FTOT * 2));
  __bf16* win_bf = (__bf16*)(ws + alloc((size_t)FTOT * FTOT * 2));
  float*  h0     = (float*)(ws + alloc((size_t)MROWS * FTOT * 4));
  __bf16* h0_bf  = (__bf16*)(ws + alloc((size_t)MROWS * FTOT * 2));
  __bf16* wc0    = (__bf16*)(ws + alloc((size_t)FTOT * FTOT * 2));
  __bf16* wc1    = (__bf16*)(ws + alloc((size_t)FTOT * FTOT * 2));
  __bf16* wc2    = (__bf16*)(ws + alloc((size_t)FTOT * FTOT * 2));
  float*  hc     = (float*)(ws + alloc((size_t)MROWS * FTOT * 4));
  float*  hp     = (float*)(ws + alloc((size_t)MROWS * FTOT * 4));
  __bf16* hpb    = (__bf16*)(ws + alloc((size_t)MROWS * FTOT * 2));
  __bf16* wB_bf  = (__bf16*)(ws + alloc((size_t)NSTATE * FTOT * 2));
  __bf16* wC_bf  = (__bf16*)(ws + alloc((size_t)NSTATE * FTOT * 2));
  __bf16* wd1_bf = (__bf16*)(ws + alloc((size_t)NDH * FTOT * 2));
  __bf16* wd2_bf = (__bf16*)(ws + alloc((size_t)FTOT * NDH * 2));
  __bf16* wout_bf= (__bf16*)(ws + alloc((size_t)FTOT * FTOT * 2));
  float*  Bm     = (float*)(ws + alloc((size_t)MROWS * NSTATE * 4));
  float*  Cm     = (float*)(ws + alloc((size_t)MROWS * NSTATE * 4));
  __bf16* t1_bf  = (__bf16*)(ws + alloc((size_t)MROWS * NDH * 2));
  float*  Delta  = (float*)(ws + alloc((size_t)MROWS * FTOT * 4));
  float*  yv     = (float*)(ws + alloc((size_t)MROWS * FTOT * 4));
  __bf16* gn     = (__bf16*)(ws + alloc((size_t)MROWS * FTOT * 2));

  // scratch stats must be zero each call (deterministic under graph replay)
  k_zero<<<1, 256, 0, stream>>>(stats, 16);
  k_zero<<<8, 256, 0, stream>>>(csum, MROWS);
  k_zero<<<8, 256, 0, stream>>>(csq, MROWS);

  // weight conversions to bf16
  k_f2b<<<4096, 256, 0, stream>>>(w_in, win_bf, FTOT * FTOT);
  k_convw<<<4096, 256, 0, stream>>>(conv_w, wc0, wc1, wc2);
  k_f2b<<<64, 256, 0, stream>>>(w_B, wB_bf, NSTATE * FTOT);
  k_f2b<<<64, 256, 0, stream>>>(w_C, wC_bf, NSTATE * FTOT);
  k_f2b<<<256, 256, 0, stream>>>(w_d1, wd1_bf, NDH * FTOT);
  k_f2b<<<256, 256, 0, stream>>>(w_d2, wd2_bf, FTOT * NDH);
  k_f2b<<<4096, 256, 0, stream>>>(w_out, wout_bf, FTOT * FTOT);

  // GroupNorm(1) + in-projection (WMMA, 16x64 strips)
  k_batch_stats<<<128, 256, 0, stream>>>(x, stats);
  k_xnorm<<<8192, 256, 0, stream>>>(x, stats, xn_bf);
  // strips = (2048/16) * (1024/64) = 2048 -> 256 blocks of 8 waves
  k_wmma_gemm<4><<<256, 256, 0, stream>>>(xn_bf, win_bf, b_in, h0, h0_bf, nullptr,
                                          MROWS, FTOT, FTOT, 4);

  // conv block (3-tap WMMA) + InstanceNorm + SiLU
  k_conv_wmma<<<256, 256, 0, stream>>>(h0_bf, wc0, wc1, wc2, hc);
  k_chan_stats<<<128, 256, 0, stream>>>(hc, csum, csq);
  k_chan_norm_silu<<<8192, 256, 0, stream>>>(hc, csum, csq, hp, hpb);

  // B, C, Delta projections (WMMA)
  // N=16: 128 tiles -> 16 blocks (NT=1)
  k_wmma_gemm<1><<<16, 256, 0, stream>>>(hpb, wB_bf, b_B, Bm, nullptr, nullptr,
                                         MROWS, NSTATE, FTOT, 0);
  k_wmma_gemm<1><<<16, 256, 0, stream>>>(hpb, wC_bf, b_C, Cm, nullptr, nullptr,
                                         MROWS, NSTATE, FTOT, 0);
  // N=64: 128 strips -> 16 blocks (NT=4)
  k_wmma_gemm<4><<<16, 256, 0, stream>>>(hpb, wd1_bf, b_d1, nullptr, t1_bf, nullptr,
                                         MROWS, NDH, FTOT, 1);
  // N=1024, K=64: 2048 strips -> 256 blocks, + softplus epilogue
  k_wmma_gemm<4><<<256, 256, 0, stream>>>(t1_bf, wd2_bf, b_d2, Delta, nullptr, nullptr,
                                          MROWS, FTOT, NDH, 2);

  // selective scan + gate + row-norm + out-projection (WMMA, +residual)
  k_ssm_scan<<<128, 256, 0, stream>>>(Delta, hp, Bm, Cm, A_log, Dvec, yv);
  k_gate_rownorm<<<2048, 256, 0, stream>>>(yv, h0, gn);
  k_wmma_gemm<4><<<256, 256, 0, stream>>>(gn, wout_bf, b_out, out, nullptr, x,
                                          MROWS, FTOT, FTOT, 3);
}
